// AttentionHead_19121194402065
// MI455X (gfx1250) — matrix operand order repaired and checked by reading, not compiled, not run on hardware
//
#include <hip/hip_runtime.h>
#include <hip/hip_bf16.h>

constexpr int DK   = 128;
constexpr int SEQL = 1024;
constexpr int NB   = 16;

typedef __attribute__((ext_vector_type(16))) _Float16 v16h;
typedef __attribute__((ext_vector_type(8)))  float    v8f;
typedef __attribute__((ext_vector_type(4)))  int      v4i;

#define WMMA_F16(a, b, c) \
  __builtin_amdgcn_wmma_f32_16x16x32_f16(false, (a), false, (b), (short)0, (c), false, false)

// A-fragment (16x32 f16, M x K), matrix row-major with leading dim ld (halves).
// Lanes 0-15: M=lane, K chunks [0..7] and [16..23]; lanes 16-31: M=lane-16, [8..15],[24..31].
static __device__ __forceinline__ v16h frag_a16(const _Float16* base, int ld, int lane) {
  const int r = lane & 15, h = lane >> 4;
  const _Float16* p = base + r * ld + h * 8;
  union { v4i i[2]; v16h f; } u;
  u.i[0] = *(const v4i*)(p);
  u.i[1] = *(const v4i*)(p + 16);
  return u.f;
}
// B-fragment (32x16 f16, K x N), stored N-major: element (k,n) at base + n*ld + k.
// Lanes 0-15: N=lane, K chunks [0..7] and [16..23]; lanes 16-31: N=lane-16, [8..15],[24..31].
static __device__ __forceinline__ v16h frag_b16(const _Float16* base, int ld, int lane) {
  const int r = lane & 15, h = lane >> 4;
  const _Float16* p = base + r * ld + h * 8;
  union { v4i i[2]; v16h f; } u;
  u.i[0] = *(const v4i*)(p);
  u.i[1] = *(const v4i*)(p + 16);
  return u.f;
}

// gfx1250 async global->LDS copy (16B), tracked by ASYNCcnt.
static __device__ __forceinline__ void async_copy_b128(const void* gsrc, void* ldst) {
  __attribute__((address_space(3))) char* lp =
      (__attribute__((address_space(3))) char*)ldst;
  asm volatile("global_load_async_to_lds_b128 %0, %1, off"
               :
               : "v"(lp), "v"(gsrc)
               : "memory");
}
static __device__ __forceinline__ void wait_async0() {
  asm volatile("s_wait_asynccnt 0x0" ::: "memory");
}

// ---------------------------------------------------------------------------
// Kernel 1: build U via the Givens-rotation scan. Column-parallel: thread t owns
// column t, rotations only mix rows a=j-1,b=j inside a column -> no barriers in
// the scan. U kept in 64KB LDS. Emits U row-major f16 (B-layout for Uxhat GEMM)
// and Vw row-major f16 (B-layout for the V GEMM).
// ---------------------------------------------------------------------------
__global__ __launch_bounds__(128) void k_build_u(const float* __restrict__ phi,
                                                 const float* __restrict__ Vw,
                                                 _Float16* __restrict__ Uf16,
                                                 _Float16* __restrict__ Vwf16) {
  __shared__ float U[DK * DK];
  const int t = threadIdx.x;
  for (int rr = 0; rr < DK; ++rr) U[rr * DK + t] = (rr == t) ? 1.f : 0.f;
  __syncthreads();
  int step = 0;
  for (int i = 1; i < DK; ++i) {
    for (int j = i; j >= 1; --j) {
      const float th = phi[step++];
      const float c = __cosf(th), s = __sinf(th);
      const float ra = U[(j - 1) * DK + t];
      const float rb = U[j * DK + t];
      U[(j - 1) * DK + t] = c * ra + s * rb;
      U[j * DK + t]       = -s * ra + c * rb;
    }
  }
  __syncthreads();
  for (int idx = t; idx < DK * DK; idx += 128) {
    Uf16[idx]  = (_Float16)U[idx];
    Vwf16[idx] = (_Float16)Vw[idx];
  }
}

// ---------------------------------------------------------------------------
// Kernel 2: per 16-token tile (one wave): norms, xhat(f16), then WMMA GEMMs:
//   Uxj[n,d] = sum_e xhat[n,e]*U[d,e]   (A=xhat, B=U row-major)
//   V[n,e]   = sum_d x[n,d]*Vw[e,d]+Vb  (A=x16,  B=Vw row-major), stored Vt[e][j]
// ---------------------------------------------------------------------------
__global__ __launch_bounds__(128) void k_prep(const float* __restrict__ x,
                                              const float* __restrict__ Vb,
                                              const _Float16* __restrict__ Uf16,
                                              const _Float16* __restrict__ Vwf16,
                                              float* __restrict__ norms,
                                              _Float16* __restrict__ xhat16,
                                              _Float16* __restrict__ Uxj16,
                                              _Float16* __restrict__ Vt16) {
  __shared__ _Float16 xhL[64 * DK];
  __shared__ _Float16 x16L[64 * DK];
  const int lane = threadIdx.x & 31;
  const int w    = threadIdx.x >> 5;
  const int r    = lane & 15, h = lane >> 4;
  const int tok16 = blockIdx.x * 64 + w * 16;
  const int myTok = tok16 + r;  // both halves handle the same token redundantly

  // --- norms + xhat ---
  float ss = 0.f;
  const float4* xr = (const float4*)(x + (size_t)myTok * DK);
  #pragma unroll 4
  for (int q = 0; q < 32; ++q) {
    float4 v = xr[q];
    ss += v.x * v.x + v.y * v.y + v.z * v.z + v.w * v.w;
  }
  const float nrm = sqrtf(ss);
  if (h == 0) norms[myTok] = nrm;
  const float rinv = 1.f / fmaxf(nrm, 1e-12f);

  const float* xrow = x + (size_t)myTok * DK;
  _Float16* xhW  = xhL  + (w * 16 + r) * DK;
  _Float16* x16W = x16L + (w * 16 + r) * DK;
  _Float16* gxh  = xhat16 + (size_t)myTok * DK;
  for (int d = h * 64; d < h * 64 + 64; ++d) {
    const float xv = xrow[d];
    const _Float16 hv = (_Float16)(xv * rinv);
    xhW[d] = hv;
    gxh[d] = hv;
    x16W[d] = (_Float16)xv;
  }
  __syncthreads();

  // --- two fused GEMMs over the 16-token tile ---
  v16h aH[4], aF[4];
  #pragma unroll
  for (int kc = 0; kc < 4; ++kc) {
    aH[kc] = frag_a16(xhL  + w * 16 * DK + kc * 32, DK, lane);
    aF[kc] = frag_a16(x16L + w * 16 * DK + kc * 32, DK, lane);
  }
  const int bidx = tok16 >> 10;
  const int sloc = tok16 & (SEQL - 1);
  const v8f zero = {};
  #pragma unroll
  for (int nt = 0; nt < 8; ++nt) {
    v8f c1 = zero, c2 = zero;
    #pragma unroll
    for (int kc = 0; kc < 4; ++kc) {
      v16h bU = frag_b16(Uf16  + (nt * 16) * DK + kc * 32, DK, lane);
      v16h bV = frag_b16(Vwf16 + (nt * 16) * DK + kc * 32, DK, lane);
      c1 = WMMA_F16(aH[kc], bU, c1);
      c2 = WMMA_F16(aF[kc], bV, c2);
    }
    const int n = nt * 16 + r;
    const float vb = Vb[n];
    #pragma unroll
    for (int p = 0; p < 8; ++p) {
      const int m = p + h * 8;
      Uxj16[(size_t)(tok16 + m) * DK + n] = (_Float16)c1[p];
      Vt16[(size_t)bidx * (DK * SEQL) + (size_t)n * SEQL + sloc + m] =
          (_Float16)(c2[p] + vb);
    }
  }
}

// ---------------------------------------------------------------------------
// Kernel 3: flash attention + residual + LayerNorm. One wave owns 16 query rows.
// j-tiles of Uxj / Vt are staged into LDS once per block via gfx1250 async
// global->LDS copies, double-buffered and overlapped with WMMA + softmax work.
// ---------------------------------------------------------------------------
__global__ __launch_bounds__(128) void k_attn(const float* __restrict__ x,
                                              const float* __restrict__ norms,
                                              const _Float16* __restrict__ xhat16,
                                              const _Float16* __restrict__ Uxj16,
                                              const _Float16* __restrict__ Vt16,
                                              const float* __restrict__ gamma,
                                              const float* __restrict__ beta,
                                              float* __restrict__ out) {
  __shared__ _Float16 UL[2][32 * DK];   // Uxj j-tile, token-major (ld = DK)
  __shared__ _Float16 VL[2][DK * 32];   // Vt  j-tile, e-major   (ld = 32)
  __shared__ _Float16 Pbuf[4][16 * 32];
  const int tid  = threadIdx.x;
  const int lane = tid & 31;
  const int w    = tid >> 5;
  const int r    = lane & 15, h = lane >> 4;
  const int b    = blockIdx.y;
  const int i0   = blockIdx.x * 64 + w * 16;
  const float rsdk = 0.08838834764831845f;  // 1/sqrt(128)

  const _Float16* xh_b = xhat16 + (size_t)b * SEQL * DK;
  const _Float16* uq_b = Uxj16  + (size_t)b * SEQL * DK;
  const _Float16* vt_b = Vt16   + (size_t)b * DK * SEQL;
  const float*    nm_b = norms  + (size_t)b * SEQL;
  const float*    x_b  = x      + (size_t)b * SEQL * DK;
  float*          o_b  = out    + (size_t)b * SEQL * DK;

  // issue async copies of one 32-wide j-tile into buffer `buf`
  auto issue_tile = [&](int j0, int buf) {
    // Uxj rows j0..j0+31 : 8KB contiguous; 128 threads x 4 x b128
    const char* usrc = (const char*)(uq_b + (size_t)j0 * DK);
    char*       udst = (char*)(&UL[buf][0]);
    #pragma unroll
    for (int k = 0; k < 4; ++k) {
      const int o = (tid + k * 128) * 16;
      async_copy_b128(usrc + o, udst + o);
    }
    // Vt slice [e][j0..j0+31] : 128 rows of 64B, thread tid owns row e=tid
    const char* vsrc = (const char*)(vt_b + (size_t)tid * SEQL + j0);
    char*       vdst = (char*)(&VL[buf][0]) + tid * 64;
    #pragma unroll
    for (int k = 0; k < 4; ++k)
      async_copy_b128(vsrc + k * 16, vdst + k * 16);
  };

  v16h aX[4];
  #pragma unroll
  for (int kc = 0; kc < 4; ++kc)
    aX[kc] = frag_a16(xh_b + (size_t)i0 * DK + kc * 32, DK, lane);

  const v8f zero = {};
  v8f acc[8];
  #pragma unroll
  for (int e = 0; e < 8; ++e) acc[e] = zero;
  float mrow[8], lrow[8];
  #pragma unroll
  for (int p = 0; p < 8; ++p) { mrow[p] = -3.0e38f; lrow[p] = 0.f; }

  _Float16* Pb = Pbuf[w];

  issue_tile(0, 0);

  for (int jt = 0; jt < SEQL / 32; ++jt) {
    const int j0  = jt * 32;
    const int cur = jt & 1;
    wait_async0();        // my wave's async copies into `cur` are done
    __syncthreads();      // everyone's copies into `cur` are done
    if (jt + 1 < SEQL / 32) issue_tile(j0 + 32, cur ^ 1);
    const _Float16* ULc = &UL[cur][0];
    const _Float16* VLc = &VL[cur][0];

    // scores: s[16x32] = xhat_i (16x128) . Uxj_j^T   (B-frags from LDS)
    v8f s0 = zero, s1 = zero;
    #pragma unroll
    for (int kc = 0; kc < 4; ++kc) {
      v16h b0 = frag_b16(ULc + kc * 32,           DK, lane);
      v16h b1 = frag_b16(ULc + 16 * DK + kc * 32, DK, lane);
      s0 = WMMA_F16(aX[kc], b0, s0);
      s1 = WMMA_F16(aX[kc], b1, s1);
    }
    const float nj0 = nm_b[j0 + r] * rsdk;
    const float nj1 = nm_b[j0 + 16 + r] * rsdk;
    float t0[8], t1[8], tm[8];
    #pragma unroll
    for (int p = 0; p < 8; ++p) {
      t0[p] = s0[p] * s0[p] * nj0;
      t1[p] = s1[p] * s1[p] * nj1;
      tm[p] = fmaxf(t0[p], t1[p]);
    }
    // row-max across the 16-lane half (lanes of a half hold the N columns)
    #pragma unroll
    for (int p = 0; p < 8; ++p) {
      #pragma unroll
      for (int msk = 1; msk <= 8; msk <<= 1)
        tm[p] = fmaxf(tm[p], __shfl_xor(tm[p], msk, 32));
    }
    float corr[8], rs[8];
    #pragma unroll
    for (int p = 0; p < 8; ++p) {
      const float mn = fmaxf(mrow[p], tm[p]);
      corr[p] = __expf(mrow[p] - mn);
      mrow[p] = mn;
      t0[p] = __expf(t0[p] - mn);
      t1[p] = __expf(t1[p] - mn);
      rs[p] = t0[p] + t1[p];
    }
    #pragma unroll
    for (int p = 0; p < 8; ++p) {
      #pragma unroll
      for (int msk = 1; msk <= 8; msk <<= 1)
        rs[p] += __shfl_xor(rs[p], msk, 32);
      lrow[p] = lrow[p] * corr[p] + rs[p];
    }
    #pragma unroll
    for (int e = 0; e < 8; ++e) {
      #pragma unroll
      for (int p = 0; p < 8; ++p) acc[e][p] *= corr[p];
    }
    // C-layout -> A-layout for P through per-wave LDS tile (16x32 f16 row-major)
    #pragma unroll
    for (int p = 0; p < 8; ++p) {
      const int m = p + h * 8;
      Pb[m * 32 + r]      = (_Float16)t0[p];
      Pb[m * 32 + 16 + r] = (_Float16)t1[p];
    }
    asm volatile("s_wait_dscnt 0x0" ::: "memory");
    __builtin_amdgcn_wave_barrier();
    v16h pA = frag_a16(Pb, 32, lane);
    asm volatile("" ::: "memory");
    __builtin_amdgcn_wave_barrier();
    // acc += P (16x32) . V_j (32x128), Vt tile in LDS is N-major (ld=32)
    #pragma unroll
    for (int e = 0; e < 8; ++e) {
      v16h vB = frag_b16(VLc + (e * 16) * 32, 32, lane);
      acc[e] = WMMA_F16(pA, vB, acc[e]);
    }
  }

  // epilogue: h = attn@V + x, then LayerNorm over the 128 features of each row
  float hv[8][8];
  #pragma unroll
  for (int p = 0; p < 8; ++p) {
    const float rl = 1.f / lrow[p];
    const int m = p + h * 8;
    #pragma unroll
    for (int e = 0; e < 8; ++e) {
      const int n = e * 16 + r;
      hv[e][p] = acc[e][p] * rl + x_b[(size_t)(i0 + m) * DK + n];
    }
  }
  float mu[8], rstd[8];
  #pragma unroll
  for (int p = 0; p < 8; ++p) {
    float sum = 0.f;
    #pragma unroll
    for (int e = 0; e < 8; ++e) sum += hv[e][p];
    #pragma unroll
    for (int msk = 1; msk <= 8; msk <<= 1) sum += __shfl_xor(sum, msk, 32);
    mu[p] = sum * (1.f / 128.f);
  }
  #pragma unroll
  for (int p = 0; p < 8; ++p) {
    float var = 0.f;
    #pragma unroll
    for (int e = 0; e < 8; ++e) {
      const float d = hv[e][p] - mu[p];
      var += d * d;
    }
    #pragma unroll
    for (int msk = 1; msk <= 8; msk <<= 1) var += __shfl_xor(var, msk, 32);
    rstd[p] = rsqrtf(var * (1.f / 128.f) + 1e-5f);
  }
  #pragma unroll
  for (int e = 0; e < 8; ++e) {
    const int n = e * 16 + r;
    const float g = gamma[n], be = beta[n];
    #pragma unroll
    for (int p = 0; p < 8; ++p) {
      const int m = p + h * 8;
      o_b[(size_t)(i0 + m) * DK + n] = (hv[e][p] - mu[p]) * rstd[p] * g + be;
    }
  }
}

// ---------------------------------------------------------------------------
extern "C" void kernel_launch(void* const* d_in, const int* in_sizes, int n_in,
                              void* d_out, int out_size, void* d_ws, size_t ws_size,
                              hipStream_t stream) {
  const float* x     = (const float*)d_in[0];
  const float* phi   = (const float*)d_in[1];
  const float* Vw    = (const float*)d_in[2];
  const float* Vb    = (const float*)d_in[3];
  const float* gamma = (const float*)d_in[4];
  const float* beta  = (const float*)d_in[5];
  float* out = (float*)d_out;

  char* ws = (char*)d_ws;
  size_t off = 0;
  _Float16* Uf16   = (_Float16*)(ws + off); off += (size_t)DK * DK * 2;            // 32 KB
  _Float16* Vwf16  = (_Float16*)(ws + off); off += (size_t)DK * DK * 2;            // 32 KB
  float*    normsW = (float*)   (ws + off); off += (size_t)NB * SEQL * 4;          // 64 KB
  _Float16* xhat16 = (_Float16*)(ws + off); off += (size_t)NB * SEQL * DK * 2;     // 4 MB
  _Float16* Uxj16  = (_Float16*)(ws + off); off += (size_t)NB * SEQL * DK * 2;     // 4 MB
  _Float16* Vt16   = (_Float16*)(ws + off); off += (size_t)NB * SEQL * DK * 2;     // 4 MB
  (void)ws_size; (void)in_sizes; (void)n_in; (void)out_size;

  k_build_u<<<1, 128, 0, stream>>>(phi, Vw, Uf16, Vwf16);
  k_prep<<<NB * SEQL / 64, 128, 0, stream>>>(x, Vb, Uf16, Vwf16, normsW, xhat16,
                                             Uxj16, Vt16);
  k_attn<<<dim3(SEQL / 64, NB), 128, 0, stream>>>(x, normsW, xhat16, Uxj16, Vt16,
                                                  gamma, beta, out);
}
